// SimpleLinearAttention_47476568490195
// MI455X (gfx1250) — compile-verified, matrix operand order checked by reading, not run on hardware
//
#include <hip/hip_runtime.h>

typedef __attribute__((ext_vector_type(4)))  __bf16 v4bf;
typedef __attribute__((ext_vector_type(8)))  __bf16 v8bf;
typedef __attribute__((ext_vector_type(16))) __bf16 v16bf;
typedef __attribute__((ext_vector_type(8)))  float  v8f;
typedef __attribute__((ext_vector_type(4)))  unsigned int u32x4;
typedef __attribute__((ext_vector_type(8)))  int i32x8;
typedef __attribute__((ext_vector_type(4)))  int i32x4;

#define TILE   128
#define TK     32
#define LDT    40          // padded LDS row stride (bf16): 80B = 16 data DWORDs + 4 pad DWORDs
#define SB     4
#define SS     4096
#define SH     1024
#define NCHUNK 64
#define CLEN   (SS / NCHUNK)

#if defined(__gfx1250__) && __has_builtin(__builtin_amdgcn_tensor_load_to_lds)
#define USE_TDM 1
#else
#define USE_TDM 0
#endif

#if USE_TDM
__device__ __forceinline__ unsigned lds_byte_offset(const void* p) {
  // generic -> LDS address space: yields the 32-bit LDS offset
  return (unsigned)(unsigned long long)(__attribute__((address_space(3))) const char*)p;
}

// Issue one TDM 2D tile load: 32 (k) x 128 (rows) bf16, row stride = strideElems,
// LDS padding 16B after every 64B row chunk -> matches LDT=40 layout.
__device__ __forceinline__ void tdm_load_tile(unsigned ldsOff, unsigned long long gaddr,
                                              int strideElems) {
  u32x4 g0;
  g0[0] = 1u;                                          // count=1 (valid user descriptor)
  g0[1] = ldsOff;                                      // lds_addr
  g0[2] = (unsigned)(gaddr & 0xFFFFFFFFull);           // global_addr[31:0]
  g0[3] = (unsigned)((gaddr >> 32) & 0x01FFFFFFull)    // global_addr[56:32]
        | (2u << 30);                                  // type = 2 (image)
  i32x8 g1;
  g1[0] = (1 << 16)                                    // data_size = 2 bytes
        | (1 << 20)                                    // pad_enable
        | (3 << 22)                                    // pad_interval: 16 DWORDs (64B)
        | (3 << 25);                                   // pad_amount: 4 DWORDs (16B)
  g1[1] = (int)((unsigned)TK << 16);                   // tensor_dim0[15:0] = 32
  g1[2] = (int)((unsigned)TILE << 16);                 // tensor_dim0[31:16]=0 | tensor_dim1[15:0]=128
  g1[3] = (int)((unsigned)TK << 16);                   // tensor_dim1[31:16]=0 | tile_dim0 = 32
  g1[4] = TILE;                                        // tile_dim1 = 128, tile_dim2 = 0
  g1[5] = strideElems;                                 // tensor_dim0_stride[31:0]
  g1[6] = 0;                                           // stride[47:32]=0 | dim1_stride lo = 0
  g1[7] = 0;
  i32x4 z4 = {0, 0, 0, 0};
#if __clang_major__ >= 23
  i32x8 z8 = {0, 0, 0, 0, 0, 0, 0, 0};
  __builtin_amdgcn_tensor_load_to_lds(g0, g1, z4, z4, z8, 0);
#else
  __builtin_amdgcn_tensor_load_to_lds(g0, g1, z4, z4, 0);
#endif
}
#endif  // USE_TDM

// ---------------------------------------------------------------------------
// fp32 -> bf16 streaming conversion (activations & weights), float4 granule.
// ---------------------------------------------------------------------------
__global__ __launch_bounds__(256) void cvt_bf16_kernel(const float* __restrict__ in,
                                                       __bf16* __restrict__ out, int n4) {
  int i = blockIdx.x * 256 + threadIdx.x;
  if (i < n4) {
    float4 f = ((const float4*)in)[i];
    v4bf v = { (__bf16)f.x, (__bf16)f.y, (__bf16)f.z, (__bf16)f.w };
    ((v4bf*)out)[i] = v;
  }
}

// ---------------------------------------------------------------------------
// GEMM: out[m,n] = sum_k A[m,k]*W[n,k] + bias[n]   (A, W pre-converted bf16)
// 128x128 block, K-step 32, 8 waves (4Mx2N), each wave 32x64 via 2x4
// v_wmma_f32_16x16x32_bf16. Double-buffered LDS; tiles staged by the Tensor
// Data Mover (wave0: A, wave1: B), overlapped with WMMA on the other buffer.
// ---------------------------------------------------------------------------
__global__ __launch_bounds__(256) void gemm_bias_kernel(
    const __bf16* __restrict__ A, const __bf16* __restrict__ W,
    const float* __restrict__ bias, float* __restrict__ out,
    int Mdim, int Ndim, int Kdim)
{
  __shared__ __bf16 As[2][TILE * LDT];
  __shared__ __bf16 Bs[2][TILE * LDT];

  const int tid    = threadIdx.x;
  const int lane   = tid & 31;
  const int wave   = tid >> 5;
  const int waveM  = (wave & 3) * 32;
  const int waveN  = (wave >> 2) * 64;
  const int blockM = blockIdx.x * TILE;
  const int blockN = blockIdx.y * TILE;
  const int laneLo = lane & 15;
  const int khalf  = lane >> 4;
  const int nIter  = Kdim / TK;

  v8f acc[2][4];
#pragma unroll
  for (int mt = 0; mt < 2; ++mt)
#pragma unroll
    for (int nt = 0; nt < 4; ++nt)
#pragma unroll
      for (int i = 0; i < 8; ++i) acc[mt][nt][i] = 0.0f;

#if USE_TDM
  // prologue: DMA first tiles into buffer 0
  if (wave == 0)
    tdm_load_tile(lds_byte_offset(&As[0][0]),
                  (unsigned long long)(const void*)(A + (size_t)blockM * Kdim), Kdim);
  else if (wave == 1)
    tdm_load_tile(lds_byte_offset(&Bs[0][0]),
                  (unsigned long long)(const void*)(W + (size_t)blockN * Kdim), Kdim);
#endif

  for (int it = 0; it < nIter; ++it) {
    const int cur = it & 1;
    const int kk  = it * TK;

#if USE_TDM
    if (it + 1 < nIter) {
      // overlap: DMA tile it+1 into the other buffer while we compute tile it
      if (wave == 0)
        tdm_load_tile(lds_byte_offset(&As[cur ^ 1][0]),
                      (unsigned long long)(const void*)(A + (size_t)blockM * Kdim + kk + TK),
                      Kdim);
      else if (wave == 1)
        tdm_load_tile(lds_byte_offset(&Bs[cur ^ 1][0]),
                      (unsigned long long)(const void*)(W + (size_t)blockN * Kdim + kk + TK),
                      Kdim);
      __builtin_amdgcn_s_wait_tensorcnt(1);   // oldest (current buffer) complete
    } else {
      __builtin_amdgcn_s_wait_tensorcnt(0);
    }
#else
    // synchronous staging fallback: 16B bf16 chunks, 512 per tile, 2 per thread
#pragma unroll
    for (int it2 = 0; it2 < 2; ++it2) {
      int chunk = tid + it2 * 256;
      int row = chunk >> 2;
      int c8  = chunk & 3;
      *(v8bf*)(&As[cur][row * LDT + c8 * 8]) =
          *(const v8bf*)(A + (size_t)(blockM + row) * Kdim + kk + c8 * 8);
      *(v8bf*)(&Bs[cur][row * LDT + c8 * 8]) =
          *(const v8bf*)(W + (size_t)(blockN + row) * Kdim + kk + c8 * 8);
    }
#endif
    __syncthreads();

    // fragments (documented 16-bit A/B VGPR layout: lanes 0-15 K0-7/16-23,
    // lanes 16-31 K8-15/24-31)
    v16bf afrag[2], bfrag[4];
#pragma unroll
    for (int mt = 0; mt < 2; ++mt) {
      int m = waveM + mt * 16 + laneLo;
      v8bf lo = *(const v8bf*)(&As[cur][m * LDT + khalf * 8]);
      v8bf hi = *(const v8bf*)(&As[cur][m * LDT + 16 + khalf * 8]);
      afrag[mt] = __builtin_shufflevector(lo, hi, 0,1,2,3,4,5,6,7,8,9,10,11,12,13,14,15);
    }
#pragma unroll
    for (int nt = 0; nt < 4; ++nt) {
      int n = waveN + nt * 16 + laneLo;
      v8bf lo = *(const v8bf*)(&Bs[cur][n * LDT + khalf * 8]);
      v8bf hi = *(const v8bf*)(&Bs[cur][n * LDT + 16 + khalf * 8]);
      bfrag[nt] = __builtin_shufflevector(lo, hi, 0,1,2,3,4,5,6,7,8,9,10,11,12,13,14,15);
    }

#pragma unroll
    for (int mt = 0; mt < 2; ++mt)
#pragma unroll
      for (int nt = 0; nt < 4; ++nt)
        acc[mt][nt] = __builtin_amdgcn_wmma_f32_16x16x32_bf16(
            false, afrag[mt], false, bfrag[nt], (short)0, acc[mt][nt], false, false);

    __syncthreads();   // protect buffer before it is DMA'd/staged again
  }

  // epilogue: documented C/D layout (VGPR i -> M=i (+8 hi lanes), N=lane&15)
#pragma unroll
  for (int mt = 0; mt < 2; ++mt) {
#pragma unroll
    for (int nt = 0; nt < 4; ++nt) {
      int col   = blockN + waveN + nt * 16 + laneLo;
      float bb  = bias[col];
      int rbase = blockM + waveM + mt * 16 + khalf * 8;
#pragma unroll
      for (int i = 0; i < 8; ++i)
        out[(size_t)(rbase + i) * Ndim + col] = acc[mt][nt][i] + bb;
    }
  }
}

// ---------------------------------------------------------------------------
// Chunked causal scan (3 passes): 262K threads instead of a 4096-thread serial
// scan; k/v (128MB) stay resident in the 192MB L2 between pass 1 and pass 3.
// ---------------------------------------------------------------------------
__global__ __launch_bounds__(256) void scan_partial_kernel(
    const float* __restrict__ k, const float* __restrict__ v,
    float* __restrict__ pk, float* __restrict__ pkv)
{
  int idx  = blockIdx.x * 256 + threadIdx.x;   // (b*NCHUNK + c)*SH + h
  int h    = idx % SH;
  int rest = idx / SH;
  int c    = rest % NCHUNK;
  int b    = rest / NCHUNK;
  size_t base = ((size_t)b * SS + (size_t)c * CLEN) * SH + h;
  float ks = 0.0f, kvs = 0.0f;
#pragma unroll 4
  for (int s = 0; s < CLEN; ++s) {
    size_t o = base + (size_t)s * SH;
    __builtin_prefetch(&k[o + 8 * SH], 0, 1);
    __builtin_prefetch(&v[o + 8 * SH], 0, 1);
    float kc = k[o];
    float vc = v[o];
    ks  += kc;
    kvs += kc * vc;
  }
  pk[idx]  = ks;
  pkv[idx] = kvs;
}

__global__ __launch_bounds__(256) void scan_prefix_kernel(
    float* __restrict__ pk, float* __restrict__ pkv)
{
  int idx = blockIdx.x * 256 + threadIdx.x;    // b*SH + h
  int h = idx % SH;
  int b = idx / SH;
  float rk = 0.0f, rkv = 0.0f;
  for (int c = 0; c < NCHUNK; ++c) {
    size_t o = ((size_t)b * NCHUNK + c) * SH + h;
    float tk = pk[o], tkv = pkv[o];
    pk[o]  = rk;                                // exclusive prefix
    pkv[o] = rkv;
    rk  += tk;
    rkv += tkv;
  }
}

__global__ __launch_bounds__(256) void scan_apply_kernel(
    const float* __restrict__ q, const float* __restrict__ k,
    const float* __restrict__ v, const float* __restrict__ pk,
    const float* __restrict__ pkv, __bf16* __restrict__ att)
{
  int idx  = blockIdx.x * 256 + threadIdx.x;
  int h    = idx % SH;
  int rest = idx / SH;
  int c    = rest % NCHUNK;
  int b    = rest / NCHUNK;
  size_t base = ((size_t)b * SS + (size_t)c * CLEN) * SH + h;
  float ks = pk[idx], kvs = pkv[idx];
#pragma unroll 4
  for (int s = 0; s < CLEN; ++s) {
    size_t o = base + (size_t)s * SH;
    __builtin_prefetch(&q[o + 8 * SH], 0, 1);
    float kc = k[o];
    float vc = v[o];
    float qc = q[o];
    ks  += kc;
    kvs += kc * vc;
    float a = kvs / (ks + 1e-8f);
    att[o] = (__bf16)(a * qc);
  }
}

// ---------------------------------------------------------------------------
// Residual + LayerNorm: one 256-thread block per H=1024 row.
// ---------------------------------------------------------------------------
__global__ __launch_bounds__(256) void ln_kernel(
    const float* __restrict__ o, const float* __restrict__ residual,
    const float* __restrict__ gamma, const float* __restrict__ beta,
    float* __restrict__ out)
{
  __shared__ float ws0[8], ws1[8];
  const size_t row = blockIdx.x;
  const int tid  = threadIdx.x;
  const int lane = tid & 31;
  const int wave = tid >> 5;
  const float* orow = o + row * SH;
  const float* rrow = residual + row * SH;

  float x[4];
  float s = 0.0f, s2 = 0.0f;
#pragma unroll
  for (int i = 0; i < 4; ++i) {
    int j = tid + i * 256;
    x[i] = orow[j] + rrow[j];
    s  += x[i];
    s2 += x[i] * x[i];
  }
#pragma unroll
  for (int off = 16; off > 0; off >>= 1) {
    s  += __shfl_down(s,  off, 32);
    s2 += __shfl_down(s2, off, 32);
  }
  if (lane == 0) { ws0[wave] = s; ws1[wave] = s2; }
  __syncthreads();
  if (tid == 0) {
    float t = 0.0f, t2 = 0.0f;
#pragma unroll
    for (int w = 0; w < 8; ++w) { t += ws0[w]; t2 += ws1[w]; }
    ws0[0] = t; ws1[0] = t2;
  }
  __syncthreads();
  const float mean = ws0[0] * (1.0f / (float)SH);
  const float var  = ws1[0] * (1.0f / (float)SH) - mean * mean;
  const float inv  = rsqrtf(var + 1e-5f);
#pragma unroll
  for (int i = 0; i < 4; ++i) {
    int j = tid + i * 256;
    out[row * SH + j] = (x[i] - mean) * inv * gamma[j] + beta[j];
  }
}

// ---------------------------------------------------------------------------
extern "C" void kernel_launch(void* const* d_in, const int* in_sizes, int n_in,
                              void* d_out, int out_size, void* d_ws, size_t ws_size,
                              hipStream_t stream) {
  const float* query = (const float*)d_in[0];
  const float* key_  = (const float*)d_in[1];
  const float* value = (const float*)d_in[2];
  const float* Wq    = (const float*)d_in[3];
  const float* bq    = (const float*)d_in[4];
  const float* Wk    = (const float*)d_in[5];
  const float* bk    = (const float*)d_in[6];
  const float* Wv    = (const float*)d_in[7];
  const float* bv    = (const float*)d_in[8];
  const float* Wo    = (const float*)d_in[9];
  const float* bo    = (const float*)d_in[10];
  const float* gamma = (const float*)d_in[11];
  const float* beta  = (const float*)d_in[12];

  const int Mdim = SB * SS;                                     // 16384
  const size_t actF32  = (size_t)Mdim * SH * sizeof(float);     // 64 MB
  const size_t actBF16 = (size_t)Mdim * SH * sizeof(__bf16);    // 32 MB
  const size_t wBF16   = (size_t)SH * SH * sizeof(__bf16);      // 2 MB

  char* ws = (char*)d_ws;
  float*  qf  = (float*)(ws);                                   // f32 q
  float*  kf  = (float*)(ws + actF32);                          // f32 k
  float*  vf  = (float*)(ws + 2 * actF32);                      // f32 v
  __bf16* qb  = (__bf16*)(ws + 3 * actF32);                     // bf16 query
  __bf16* kb  = (__bf16*)(ws + 3 * actF32 + actBF16);           // bf16 key
  __bf16* vb  = (__bf16*)(ws + 3 * actF32 + 2 * actBF16);       // bf16 value
  __bf16* wqb = (__bf16*)(ws + 3 * actF32 + 3 * actBF16);
  __bf16* wkb = (__bf16*)((char*)wqb + wBF16);
  __bf16* wvb = (__bf16*)((char*)wkb + wBF16);
  __bf16* wob = (__bf16*)((char*)wvb + wBF16);
  __bf16* att = qb;                                             // reuse (qb dead after GEMM-q)
  float*  pk  = (float*)kb;                                     // reuse (kb dead after GEMM-k)
  float*  pkv = (float*)((char*)kb + (size_t)SB * NCHUNK * SH * 4);
  float*  of  = qf;                                             // reuse (qf dead after scan)

  const int act4 = Mdim * SH / 4, w4 = SH * SH / 4;
  cvt_bf16_kernel<<<act4 / 256, 256, 0, stream>>>(query, qb, act4);
  cvt_bf16_kernel<<<act4 / 256, 256, 0, stream>>>(key_,  kb, act4);
  cvt_bf16_kernel<<<act4 / 256, 256, 0, stream>>>(value, vb, act4);
  cvt_bf16_kernel<<<w4 / 256, 256, 0, stream>>>(Wq, wqb, w4);
  cvt_bf16_kernel<<<w4 / 256, 256, 0, stream>>>(Wk, wkb, w4);
  cvt_bf16_kernel<<<w4 / 256, 256, 0, stream>>>(Wv, wvb, w4);
  cvt_bf16_kernel<<<w4 / 256, 256, 0, stream>>>(Wo, wob, w4);

  dim3 ggrid(Mdim / TILE, SH / TILE);                           // (128, 8)
  gemm_bias_kernel<<<ggrid, 256, 0, stream>>>(qb, wqb, bq, qf, Mdim, SH, SH);
  gemm_bias_kernel<<<ggrid, 256, 0, stream>>>(kb, wkb, bk, kf, Mdim, SH, SH);
  gemm_bias_kernel<<<ggrid, 256, 0, stream>>>(vb, wvb, bv, vf, Mdim, SH, SH);

  scan_partial_kernel<<<(SB * SH * NCHUNK) / 256, 256, 0, stream>>>(kf, vf, pk, pkv);
  scan_prefix_kernel<<<(SB * SH) / 256, 256, 0, stream>>>(pk, pkv);
  scan_apply_kernel<<<(SB * SH * NCHUNK) / 256, 256, 0, stream>>>(qf, kf, vf, pk, pkv, att);

  gemm_bias_kernel<<<ggrid, 256, 0, stream>>>(att, wob, bo, of, Mdim, SH, SH);

  ln_kernel<<<Mdim, 256, 0, stream>>>(of, query, gamma, beta, (float*)d_out);
}
